// LSTM_73744588473077
// MI455X (gfx1250) — matrix operand order repaired and checked by reading, not compiled, not run on hardware
//
#include <hip/hip_runtime.h>
#include <hip/hip_bf16.h>
#include <math.h>

typedef __attribute__((ext_vector_type(16))) _Float16 v16h;
typedef __attribute__((ext_vector_type(8)))  _Float16 v8h;
typedef __attribute__((ext_vector_type(8)))  float    v8f;

#define UNITS 50
#define FEAT  128
#define OUT_N 74
#define T_LEN 256
#define B_ALL 1024
#define G4    (4*UNITS)      // 200 gate columns
#define NT    13             // ceil(200/16) N tiles
#define XH_STRIDE 192        // 128 x-feats + 64 padded h (K total = 192 = 6 chunks of 32)
#define Z_STRIDE  208

__device__ __forceinline__ float sigf(float x) { return 1.0f / (1.0f + __expf(-x)); }

// One workgroup = 16 batch rows x 1 direction. 8 waves; wave w owns gate
// column tiles {w, w+8 (if <13)}. Weights ([W;U] as f16 B-fragments) are held
// in registers for the entire T=256 recurrence; x is software-pipelined one
// timestep ahead in registers. Two barriers per timestep: x-staging and the
// h-update write disjoint LDS regions, so they share one phase.
__global__ __launch_bounds__(256) void bilstm_kernel(
    const float* __restrict__ x,
    const float* __restrict__ W_f, const float* __restrict__ U_f, const float* __restrict__ b_f,
    const float* __restrict__ W_b, const float* __restrict__ U_b, const float* __restrict__ b_b,
    float* __restrict__ h_all)
{
  const int dir  = blockIdx.x & 1;
  const int bt   = blockIdx.x >> 1;
  const int b0   = bt * 16;
  const int tid  = threadIdx.x;
  const int wave = tid >> 5;
  const int lane = tid & 31;
  const int ln16 = lane & 15;
  const int hs   = lane >> 4;      // half-wave select

  // Scalar (SGPR) wave id -> uniform branches, no exec-mask churn around WMMA.
  const int  wave_s = __builtin_amdgcn_readfirstlane(wave);
  const int  nt0    = wave;
  const int  nt1    = wave + 8;
  const bool has1   = (wave_s + 8) < NT;   // scalar: waves 0..4 own a 2nd tile

  const float* W    = dir ? W_b : W_f;
  const float* U    = dir ? U_b : U_f;
  const float* bias = dir ? b_b : b_f;

  __shared__ _Float16 xh[16 * XH_STRIDE];  // A staging: [m][k], f16, 6144 B
  __shared__ float    zs[16 * Z_STRIDE];   // gate pre-activations, 13312 B
  __shared__ float    cs[16 * 52];         // cell state, 3328 B

  // ---- load combined [W;U;0] (192 x 208) B-fragments into registers, once ----
  // 16-bit B layout (col n = lane&15): halves 0..7 -> K = 8hs + i ; halves 8..15 -> +16.
  v16h  bfr[2][6];
  float bias_v[2];
  #pragma unroll
  for (int p = 0; p < 2; ++p) {
    const int  nt = p ? nt1 : nt0;
    const int  g  = nt * 16 + ln16;
    const bool gv = (p == 0 || has1) && (g < G4);
    bias_v[p] = gv ? bias[g] : 0.0f;
    #pragma unroll
    for (int kc = 0; kc < 6; ++kc) {
      v16h bf;
      #pragma unroll
      for (int i = 0; i < 16; ++i) {
        const int k = kc * 32 + hs * 8 + (i < 8 ? i : i + 8);
        float w = 0.0f;
        if (gv) {
          if (k < FEAT)              w = W[k * G4 + g];
          else if (k < FEAT + UNITS) w = U[(k - FEAT) * G4 + g];
        }
        bf[i] = (_Float16)w;
      }
      bfr[p][kc] = bf;
    }
  }

  // ---- precompute elementwise work items (800 = 16*50 over 256 threads) ----
  const int  ei3 = tid + 768;
  const int  em0 = tid / UNITS,         eu0 = tid - em0 * UNITS;
  const int  em1 = (tid + 256) / UNITS, eu1 = (tid + 256) - em1 * UNITS;
  const int  em2 = (tid + 512) / UNITS, eu2 = (tid + 512) - em2 * UNITS;
  const int  em3 = ei3 / UNITS,         eu3 = ei3 - em3 * UNITS;
  const bool ev3 = ei3 < 16 * UNITS;

  const int xrow = tid >> 4;         // batch row within tile (16 threads per row)
  const int xcol = (tid & 15) * 8;   // 8 floats per thread

  // ---- preamble: stage x(t=0), prefetch x(t=1), zero h region + cell state ----
  const float* px = x + ((long)(b0 + xrow) * T_LEN + (dir ? T_LEN - 1 : 0)) * FEAT + xcol;
  const long   pstep = dir ? -(long)FEAT : (long)FEAT;

  float4 f0 = *(const float4*)(px);
  float4 f1 = *(const float4*)(px + 4);
  {
    v8h hv;
    hv[0] = (_Float16)f0.x; hv[1] = (_Float16)f0.y;
    hv[2] = (_Float16)f0.z; hv[3] = (_Float16)f0.w;
    hv[4] = (_Float16)f1.x; hv[5] = (_Float16)f1.y;
    hv[6] = (_Float16)f1.z; hv[7] = (_Float16)f1.w;
    *(v8h*)&xh[xrow * XH_STRIDE + xcol] = hv;
  }
  px += pstep;
  f0 = *(const float4*)(px);       // x(t=1)
  f1 = *(const float4*)(px + 4);
  px += pstep;

  for (int idx = tid; idx < 16 * 64; idx += 256)
    xh[(idx >> 6) * XH_STRIDE + FEAT + (idx & 63)] = (_Float16)0.0f;
  for (int idx = tid; idx < 16 * 52; idx += 256)
    cs[idx] = 0.0f;

  for (int t = 0; t < T_LEN; ++t) {
    __syncthreads();   // A: xh (x cols from stage, h cols from prev update) ready

    // ---- z = [x_t, h] @ [W;U] + b  (f16 WMMA, f32 accumulate) ----
    // Burst-load all 6 A chunks, then run the WMMA chains.
    const _Float16* arow = &xh[ln16 * XH_STRIDE];
    v16h afr[6];
    #pragma unroll
    for (int kc = 0; kc < 6; ++kc) {
      // 16-bit A layout: halves 0..7 -> K = 32kc + 8hs + i ; halves 8..15 -> +16
      const int base = kc * 32 + hs * 8;
      v8h lo = *(const v8h*)(arow + base);
      v8h hi = *(const v8h*)(arow + base + 16);
      afr[kc] = __builtin_shufflevector(lo, hi, 0,1,2,3,4,5,6,7,8,9,10,11,12,13,14,15);
    }

    v8f acc0, acc1;
    #pragma unroll
    for (int r = 0; r < 8; ++r) { acc0[r] = bias_v[0]; acc1[r] = bias_v[1]; }

    #pragma unroll
    for (int kc = 0; kc < 6; ++kc)
      acc0 = __builtin_amdgcn_wmma_f32_16x16x32_f16(false, afr[kc], false, bfr[0][kc],
                                                    (short)0, acc0, false, false);
    if (has1) {                        // scalar branch, EXEC untouched
      #pragma unroll
      for (int kc = 0; kc < 6; ++kc)
        acc1 = __builtin_amdgcn_wmma_f32_16x16x32_f16(false, afr[kc], false, bfr[1][kc],
                                                      (short)0, acc1, false, false);
    }

    // D layout: lane col = lane&15, VGPR r -> row r + 8*hs
    #pragma unroll
    for (int r = 0; r < 8; ++r)
      zs[(r + 8 * hs) * Z_STRIDE + nt0 * 16 + ln16] = acc0[r];
    if (has1) {
      #pragma unroll
      for (int r = 0; r < 8; ++r)
        zs[(r + 8 * hs) * Z_STRIDE + nt1 * 16 + ln16] = acc1[r];
    }
    __syncthreads();   // B: zs ready

    // ---- gate nonlinearities + state update (writes xh cols 128..191) ----
    const bool last = (t == T_LEN - 1);
    {
      #define GATE_STEP(MM, UU)                                                \
      {                                                                        \
        const int   mm = (MM), u = (UU);                                       \
        const float zi = zs[mm * Z_STRIDE + u];                                \
        const float zf = zs[mm * Z_STRIDE + UNITS + u];                        \
        const float zg = zs[mm * Z_STRIDE + 2 * UNITS + u];                    \
        const float zo = zs[mm * Z_STRIDE + 3 * UNITS + u];                    \
        float c = cs[mm * 52 + u];                                             \
        c = sigf(zf) * c + sigf(zi) * fmaxf(zg, 0.0f);                         \
        const float h = sigf(zo) * fmaxf(c, 0.0f);                             \
        cs[mm * 52 + u] = c;                                                   \
        xh[mm * XH_STRIDE + FEAT + u] = (_Float16)h;                           \
        if (last)                                                              \
          h_all[(long)(b0 + mm) * (2 * UNITS) + dir * UNITS + u] = h;          \
      }
      GATE_STEP(em0, eu0)
      GATE_STEP(em1, eu1)
      GATE_STEP(em2, eu2)
      if (ev3) GATE_STEP(em3, eu3)
      #undef GATE_STEP
    }

    // ---- stage x(t+1) (writes xh cols 0..127, same phase as h-update) ----
    if (t + 1 < T_LEN) {               // uniform
      v8h hv;
      hv[0] = (_Float16)f0.x; hv[1] = (_Float16)f0.y;
      hv[2] = (_Float16)f0.z; hv[3] = (_Float16)f0.w;
      hv[4] = (_Float16)f1.x; hv[5] = (_Float16)f1.y;
      hv[6] = (_Float16)f1.z; hv[7] = (_Float16)f1.w;
      *(v8h*)&xh[xrow * XH_STRIDE + xcol] = hv;
    }
    if (t + 2 < T_LEN) {               // uniform; guarded to stay in-bounds
      f0 = *(const float4*)(px);       // x(t+2): a full iteration of latency cover
      f1 = *(const float4*)(px + 4);
      px += pstep;
    }
  }
}

// Dense head + softmax: one block per batch row (15 MFLOP total, trivial).
__global__ __launch_bounds__(128) void head_kernel(
    const float* __restrict__ h_all, const float* __restrict__ W_d,
    const float* __restrict__ b_d, float* __restrict__ out)
{
  const int b   = blockIdx.x;
  const int tid = threadIdx.x;
  __shared__ float h[2 * UNITS];
  __shared__ float lg[OUT_N];
  __shared__ float red[2];
  if (tid < 2 * UNITS) h[tid] = h_all[(long)b * (2 * UNITS) + tid];
  __syncthreads();
  if (tid < OUT_N) {
    float s = b_d[tid];
    #pragma unroll 4
    for (int k = 0; k < 2 * UNITS; ++k) s += h[k] * W_d[k * OUT_N + tid];
    lg[tid] = s;
  }
  __syncthreads();
  if (tid == 0) {
    float m = -3.4e38f;
    for (int i = 0; i < OUT_N; ++i) m = fmaxf(m, lg[i]);
    red[0] = m;
  }
  __syncthreads();
  float e = 0.0f;
  if (tid < OUT_N) { e = __expf(lg[tid] - red[0]); lg[tid] = e; }
  __syncthreads();
  if (tid == 0) {
    float s = 0.0f;
    for (int i = 0; i < OUT_N; ++i) s += lg[i];
    red[1] = 1.0f / s;
  }
  __syncthreads();
  if (tid < OUT_N) out[(long)b * OUT_N + tid] = e * red[1];
}

extern "C" void kernel_launch(void* const* d_in, const int* in_sizes, int n_in,
                              void* d_out, int out_size, void* d_ws, size_t ws_size,
                              hipStream_t stream)
{
  (void)in_sizes; (void)n_in; (void)out_size; (void)ws_size;
  const float* x   = (const float*)d_in[0];
  const float* W_f = (const float*)d_in[1];
  const float* U_f = (const float*)d_in[2];
  const float* b_f = (const float*)d_in[3];
  const float* W_b = (const float*)d_in[4];
  const float* U_b = (const float*)d_in[5];
  const float* b_b = (const float*)d_in[6];
  const float* W_d = (const float*)d_in[7];
  const float* b_d = (const float*)d_in[8];
  float* out   = (float*)d_out;
  float* h_all = (float*)d_ws;   // 1024 * 100 floats = 400 KB scratch

  // 64 batch tiles x 2 directions, 256 threads (8 wave32) each
  bilstm_kernel<<<dim3(128), dim3(256), 0, stream>>>(x, W_f, U_f, b_f, W_b, U_b, b_b, h_all);
  head_kernel<<<dim3(B_ALL), dim3(128), 0, stream>>>(h_all, W_d, b_d, out);
}
